// MixtureOfRanksLayer_1821066133995
// MI455X (gfx1250) — matrix operand order repaired and checked by reading, not compiled, not run on hardware
//
#include <hip/hip_runtime.h>
#include <hip/hip_bf16.h>
#include <stdint.h>

#define E_ 8
#define D_ 2048
#define H_ 8192
#define R_ 64
#define N_ 4096
#define MT 128   // tokens per block

typedef __attribute__((ext_vector_type(16))) __bf16        v16bf;
typedef __attribute__((ext_vector_type(8)))  float         v8f;
typedef __attribute__((ext_vector_type(4)))  unsigned int  u32x4;
typedef __attribute__((ext_vector_type(4)))  int           v4i;

union FragU { v16bf v; u32x4 q[2]; };

// A fragment (16x32 bf16, row-major source, stride ld elements).
// ISA layout: lane holds row M=lane%16; halfwords j: K = j + 8*((j>=8)+(lane>=16))
static __device__ __forceinline__ v16bf load_a_frag(const __bf16* base, int ld, int k0, int lane) {
  const int m = lane & 15;
  const __bf16* p = base + (size_t)m * ld + k0 + ((lane >> 4) << 3);
  FragU f;
  f.q[0] = *(const u32x4*)(p);        // K = kbase + 0..7
  f.q[1] = *(const u32x4*)(p + 16);   // K = kbase + 16..23
  return f.v;
}

// B fragment (32x16 bf16) from a TRANSPOSED operand bt: [Ncols][K] row-major.
// ISA layout: lane holds column N=lane%16; halfwords j: K = j + 8*((j>=8)+(lane>=16))
static __device__ __forceinline__ v16bf load_b_frag(const __bf16* bt, int ldbt, int k0, int n0, int lane) {
  const __bf16* p = bt + (size_t)(n0 + (lane & 15)) * ldbt + k0 + ((lane >> 4) << 3);
  FragU f;
  f.q[0] = *(const u32x4*)(p);
  f.q[1] = *(const u32x4*)(p + 16);
  return f.v;
}

static __device__ __forceinline__ v8f wmma_bf16(v16bf a, v16bf b, v8f c) {
  return __builtin_amdgcn_wmma_f32_16x16x32_bf16(false, a, false, b, (short)0, c, false, false);
}

// ---- CDNA5 async global->LDS copy (ASYNCcnt), 16 bytes per lane ----
static __device__ __forceinline__ void stage16(const __bf16* gsrc, __bf16* ldst) {
#if __has_builtin(__builtin_amdgcn_global_load_async_to_lds_b128)
  __builtin_amdgcn_global_load_async_to_lds_b128(
      (__attribute__((address_space(1))) v4i*)gsrc,
      (__attribute__((address_space(3))) v4i*)ldst, 0, 0);
#else
  const unsigned loff =
      (unsigned)(uintptr_t)(__attribute__((address_space(3))) __bf16*)ldst;
  asm volatile("global_load_async_to_lds_b128 %0, %1, off"
               :: "v"(loff), "v"((unsigned long long)(uintptr_t)gsrc)
               : "memory");
#endif
}

static __device__ __forceinline__ void async_wait0() {
#if __has_builtin(__builtin_amdgcn_s_wait_asynccnt)
  __builtin_amdgcn_s_wait_asynccnt(0);
#else
  asm volatile("s_wait_asynccnt 0" ::: "memory");
#endif
}

// Copy a 64x64 bf16 region (global row stride ldg elems) -> packed LDS [64][64].
// 256 threads, each issues 2 async b128 ops.
static __device__ __forceinline__ void stage_64x64(const __bf16* g, int ldg, __bf16* l, int tid) {
  const int r = tid >> 3;          // 0..31
  const int c = (tid & 7) << 3;    // 0..56, 8-elem (16B) granules
  stage16(g + (size_t)r * ldg + c,        l + r * 64 + c);
  stage16(g + (size_t)(r + 32) * ldg + c, l + (r + 32) * 64 + c);
}

// ---------------- helper kernels ----------------
__global__ void cvt_bf16(const float* __restrict__ s, __bf16* __restrict__ d, size_t n) {
  for (size_t i = blockIdx.x * (size_t)blockDim.x + threadIdx.x; i < n;
       i += (size_t)gridDim.x * blockDim.x)
    d[i] = (__bf16)s[i];
}

// src [E][I][J] f32 -> dst [E][J][I] bf16
__global__ void tcvt_bf16(const float* __restrict__ s, __bf16* __restrict__ d, int I, int J) {
  const size_t plane = (size_t)I * J;
  const size_t total = (size_t)E_ * plane;
  for (size_t idx = blockIdx.x * (size_t)blockDim.x + threadIdx.x; idx < total;
       idx += (size_t)gridDim.x * blockDim.x) {
    size_t e = idx / plane, rem = idx - e * plane;
    size_t i = rem / (size_t)J, j = rem - i * (size_t)J;
    d[e * plane + j * (size_t)I + i] = (__bf16)s[idx];
  }
}

__global__ void zero_f32(float* __restrict__ p, size_t n) {
  for (size_t i = blockIdx.x * (size_t)blockDim.x + threadIdx.x; i < n;
       i += (size_t)gridDim.x * blockDim.x)
    p[i] = 0.0f;
}

// One wave per token: logits -> top-2 renormalized softmax weights w[N][E]
__global__ __launch_bounds__(256) void gate_kernel(const float* __restrict__ x,
                                                   const float* __restrict__ gw,
                                                   const float* __restrict__ gb,
                                                   float* __restrict__ w) {
  const int token = blockIdx.x * (blockDim.x >> 5) + (threadIdx.x >> 5);
  const int lane  = threadIdx.x & 31;
  if (token >= N_) return;
  float acc[E_];
#pragma unroll
  for (int e = 0; e < E_; ++e) acc[e] = 0.0f;
  const float* xr = x + (size_t)token * D_;
  for (int d = lane; d < D_; d += 32) {
    const float xv = xr[d];
#pragma unroll
    for (int e = 0; e < E_; ++e) acc[e] += xv * gw[(size_t)e * D_ + d];
  }
#pragma unroll
  for (int e = 0; e < E_; ++e)
    for (int off = 16; off > 0; off >>= 1) acc[e] += __shfl_xor(acc[e], off, 32);
  if (lane == 0) {
    float lg[E_];
#pragma unroll
    for (int e = 0; e < E_; ++e) lg[e] = acc[e] + gb[e];
    int i1 = 0;
    for (int e = 1; e < E_; ++e) if (lg[e] > lg[i1]) i1 = e;
    int i2 = (i1 == 0) ? 1 : 0;
    for (int e = 0; e < E_; ++e) if (e != i1 && lg[e] > lg[i2]) i2 = e;
    const float p2 = __expf(lg[i2] - lg[i1]);
    const float inv = 1.0f / (1.0f + p2);
    float* wr = w + (size_t)token * E_;
#pragma unroll
    for (int e = 0; e < E_; ++e) wr[e] = 0.0f;
    wr[i1] = inv;
    wr[i2] = p2 * inv;
  }
}

// ---------------- fused low-rank expert kernel ----------------
// grid: (N/MT, E), block: 256 (8 waves). Wave w owns token rows [w*16, w*16+16).
// All shared B-operand 64x64 chunks are async-staged into LDS once per block
// (double-buffered), eliminating the 8x per-wave L2 read duplication.
__global__ __launch_bounds__(256) void moe_expert_kernel(
    const __bf16* __restrict__ xb,  // [N][D]
    const __bf16* __restrict__ u1t, // [E][R][D]
    const __bf16* __restrict__ v1t, // [E][H][R]
    const __bf16* __restrict__ u2t, // [E][R][H]
    const __bf16* __restrict__ v2t, // [E][D][R]
    const float* __restrict__ b1,   // [E][H]
    const float* __restrict__ b2,   // [E][D]
    const float* __restrict__ wg,   // [N][E]
    float* __restrict__ out)        // [N][D], atomic accumulate
{
  __shared__ __attribute__((aligned(16))) __bf16 ldsT[MT * R_];    // T1, later S (16 KB)
  __shared__ __attribute__((aligned(16))) __bf16 ldsH[MT * 64];    // H chunk    (16 KB)
  __shared__ __attribute__((aligned(16))) __bf16 bufV[2][64 * 64]; // staged B   (16 KB)
  __shared__ __attribute__((aligned(16))) __bf16 bufU[2][64 * 64]; // staged B   (16 KB)
  __shared__ float ldsW[MT];

  const int m0   = blockIdx.x * MT;
  const int e    = blockIdx.y;
  const int tid  = threadIdx.x;
  const int wave = tid >> 5;
  const int lane = tid & 31;

  if (tid < MT) ldsW[tid] = wg[(size_t)(m0 + tid) * E_ + e];
  __syncthreads();

  const __bf16* U1 = u1t + (size_t)e * R_ * D_;
  const __bf16* V1 = v1t + (size_t)e * H_ * R_;
  const __bf16* U2 = u2t + (size_t)e * R_ * H_;
  const __bf16* V2 = v2t + (size_t)e * D_ * R_;

  const int row8 = wave * 16 + ((lane >> 4) << 3);  // C-frag base row (absolute in tile)
  const int ncol = lane & 15;                        // C-frag column within 16
  const __bf16* trow = ldsT + wave * 16 * R_;
  const __bf16* hrow = ldsH + wave * 16 * 64;

  // ---- Stage A: T1[128x64] = X_tile @ U1e, K = 2048 in 64-wide staged chunks ----
  {
    v8f acc[4];
#pragma unroll
    for (int nt = 0; nt < 4; ++nt) acc[nt] = {0.f,0.f,0.f,0.f,0.f,0.f,0.f,0.f};
    const __bf16* arow = xb + (size_t)(m0 + wave * 16) * D_;
    stage_64x64(U1, D_, bufV[0], tid);  // U1T cols kc..kc+63, all 64 rows
    for (int kc = 0, it = 0; kc < D_; kc += 64, ++it) {
      async_wait0();
      __syncthreads();
      if (kc + 64 < D_) stage_64x64(U1 + kc + 64, D_, bufV[(it + 1) & 1], tid);
      const __bf16* bb = bufV[it & 1];
#pragma unroll
      for (int k0 = 0; k0 < 64; k0 += 32) {
        v16bf a = load_a_frag(arow, D_, kc + k0, lane);
#pragma unroll
        for (int nt = 0; nt < 4; ++nt)
          acc[nt] = wmma_bf16(a, load_b_frag(bb, 64, k0, nt * 16, lane), acc[nt]);
      }
    }
#pragma unroll
    for (int nt = 0; nt < 4; ++nt)
#pragma unroll
      for (int r = 0; r < 8; ++r)
        ldsT[(size_t)(row8 + r) * R_ + nt * 16 + ncol] = (__bf16)acc[nt][r];
  }
  __builtin_amdgcn_wave_barrier();  // ldsT/ldsH tiles are wave-private

  // ---- Stages B+C: stream H in 64-chunks; S[16x64] per wave stays in regs ----
  v8f accS[4];
#pragma unroll
  for (int nt = 0; nt < 4; ++nt) accS[nt] = {0.f,0.f,0.f,0.f,0.f,0.f,0.f,0.f};

  stage_64x64(V1, 64, bufV[0], tid);  // V1T rows h..h+63 (contiguous slab)
  stage_64x64(U2, H_, bufU[0], tid);  // U2T cols h..h+63, all 64 rows
  for (int h0 = 0, it = 0; h0 < H_; h0 += 64, ++it) {
    async_wait0();
    __syncthreads();
    if (h0 + 64 < H_) {
      stage_64x64(V1 + (size_t)(h0 + 64) * 64, 64, bufV[(it + 1) & 1], tid);
      stage_64x64(U2 + (h0 + 64), H_, bufU[(it + 1) & 1], tid);
    }
    const __bf16* bv = bufV[it & 1];
    const __bf16* bu = bufU[it & 1];
    // Hc = relu(T1 @ V1[:, h0:h0+64] + b1)
    v8f accB[4];
#pragma unroll
    for (int nt = 0; nt < 4; ++nt) accB[nt] = {0.f,0.f,0.f,0.f,0.f,0.f,0.f,0.f};
#pragma unroll
    for (int k0 = 0; k0 < 64; k0 += 32) {
      v16bf a = load_a_frag(trow, R_, k0, lane);
#pragma unroll
      for (int nt = 0; nt < 4; ++nt)
        accB[nt] = wmma_bf16(a, load_b_frag(bv, 64, k0, nt * 16, lane), accB[nt]);
    }
#pragma unroll
    for (int nt = 0; nt < 4; ++nt) {
      const float bias = b1[(size_t)e * H_ + h0 + nt * 16 + ncol];
#pragma unroll
      for (int r = 0; r < 8; ++r) {
        float v = accB[nt][r] + bias;
        v = v > 0.0f ? v : 0.0f;
        ldsH[(size_t)(row8 + r) * 64 + nt * 16 + ncol] = (__bf16)v;
      }
    }
    __builtin_amdgcn_wave_barrier();  // DS in-order per wave; block compiler reordering
    // S += Hc @ U2[h0:h0+64, :]
#pragma unroll
    for (int k0 = 0; k0 < 64; k0 += 32) {
      v16bf a = load_a_frag(hrow, 64, k0, lane);
#pragma unroll
      for (int nt = 0; nt < 4; ++nt)
        accS[nt] = wmma_bf16(a, load_b_frag(bu, 64, k0, nt * 16, lane), accS[nt]);
    }
  }

  // ---- store S (bf16) into ldsT (reuse) in A-fragment layout ----
#pragma unroll
  for (int nt = 0; nt < 4; ++nt)
#pragma unroll
    for (int r = 0; r < 8; ++r)
      ldsT[(size_t)(row8 + r) * R_ + nt * 16 + ncol] = (__bf16)accS[nt][r];
  __builtin_amdgcn_wave_barrier();

  // ---- Stage D: Y = S @ V2e + b2; out += w_e * Y, N = 2048 in 64-wide chunks ----
  {
    v16bf a0 = load_a_frag(trow, R_, 0, lane);
    v16bf a1 = load_a_frag(trow, R_, 32, lane);
    float wv[8];
#pragma unroll
    for (int r = 0; r < 8; ++r) wv[r] = ldsW[row8 + r];
    stage_64x64(V2, 64, bufV[0], tid);  // V2T rows n..n+63 (contiguous slab)
    for (int nc = 0, it = 0; nc < D_; nc += 64, ++it) {
      async_wait0();
      __syncthreads();
      if (nc + 64 < D_) stage_64x64(V2 + (size_t)(nc + 64) * 64, 64, bufV[(it + 1) & 1], tid);
      const __bf16* bv = bufV[it & 1];
#pragma unroll
      for (int ntl = 0; ntl < 4; ++ntl) {
        v8f acc = {0.f,0.f,0.f,0.f,0.f,0.f,0.f,0.f};
        acc = wmma_bf16(a0, load_b_frag(bv, 64, 0,  ntl * 16, lane), acc);
        acc = wmma_bf16(a1, load_b_frag(bv, 64, 32, ntl * 16, lane), acc);
        const int col = nc + ntl * 16 + ncol;
        const float bias = b2[(size_t)e * D_ + col];
#pragma unroll
        for (int r = 0; r < 8; ++r) {
          const float val = wv[r] * (acc[r] + bias);
          atomicAdd(&out[(size_t)(m0 + row8 + r) * D_ + col], val);
        }
      }
    }
  }
}

extern "C" void kernel_launch(void* const* d_in, const int* in_sizes, int n_in,
                              void* d_out, int out_size, void* d_ws, size_t ws_size,
                              hipStream_t stream) {
  const float* x  = (const float*)d_in[0];
  const float* u1 = (const float*)d_in[1];
  const float* v1 = (const float*)d_in[2];
  const float* b1 = (const float*)d_in[3];
  const float* u2 = (const float*)d_in[4];
  const float* v2 = (const float*)d_in[5];
  const float* b2 = (const float*)d_in[6];
  const float* gw = (const float*)d_in[7];
  const float* gb = (const float*)d_in[8];
  float* out = (float*)d_out;

  char* ws = (char*)d_ws;
  size_t off = 0;
  auto alloc = [&](size_t bytes) -> char* {
    char* p = ws + off;
    off += (bytes + 255) & ~(size_t)255;
    return p;
  };
  __bf16* xb   = (__bf16*)alloc((size_t)N_ * D_ * 2);  // 16 MB
  __bf16* u1t  = (__bf16*)alloc((size_t)E_ * R_ * D_ * 2);
  __bf16* v1t  = (__bf16*)alloc((size_t)E_ * H_ * R_ * 2);
  __bf16* u2t  = (__bf16*)alloc((size_t)E_ * R_ * H_ * 2);
  __bf16* v2t  = (__bf16*)alloc((size_t)E_ * D_ * R_ * 2);
  float*  wgt  = (float*)alloc((size_t)N_ * E_ * 4);

  // 1) f32 -> bf16 conversions (+ transposes so B-fragments read contiguous K)
  cvt_bf16<<<8192, 256, 0, stream>>>(x, xb, (size_t)N_ * D_);
  tcvt_bf16<<<4096, 256, 0, stream>>>(u1, u1t, D_, R_);  // [E][D][R] -> [E][R][D]
  tcvt_bf16<<<8192, 256, 0, stream>>>(v1, v1t, R_, H_);  // [E][R][H] -> [E][H][R]
  tcvt_bf16<<<8192, 256, 0, stream>>>(u2, u2t, H_, R_);  // [E][H][R] -> [E][R][H]
  tcvt_bf16<<<4096, 256, 0, stream>>>(v2, v2t, R_, D_);  // [E][R][D] -> [E][D][R]

  // 2) gate: top-2 softmax weights
  gate_kernel<<<N_ / 8, 256, 0, stream>>>(x, gw, gb, wgt);

  // 3) zero output, then fused experts with atomic accumulation
  zero_f32<<<8192, 256, 0, stream>>>(out, (size_t)N_ * D_);
  dim3 grid(N_ / MT, E_);
  moe_expert_kernel<<<grid, 256, 0, stream>>>(xb, u1t, v1t, u2t, v2t, b1, b2, wgt, out);
}
